// LocalAttention_86517821216559
// MI455X (gfx1250) — matrix operand order repaired and checked by reading, not compiled, not run on hardware
//
#include <hip/hip_runtime.h>
#include <hip/hip_bf16.h>
#include <stdint.h>

typedef __attribute__((ext_vector_type(16))) __bf16 v16bf;
typedef __attribute__((ext_vector_type(8)))  __bf16 v8bf;
typedef __attribute__((ext_vector_type(8)))  float  v8f;

#define B_NUM 2
#define S_LEN 2048
#define D_DIM 512
#define H_NUM 8
#define DH_DIM 64
#define LOCAL_R 7

// ---------------------------------------------------------------------------
// WMMA fragment loaders (contiguous-K order used by this code, not AMD's table order, 16-bit A 16x32 / B 32x16)
// A: lane = (half<<4)|r, row M=r. elems 0..15 -> K = half*16+0..15
//    (two 16B contiguous chunks)
// B: lane = (half<<4)|c, col N=c. elems 0..15 -> K = half*16+0..15 (32B contiguous)
// ---------------------------------------------------------------------------
__device__ __forceinline__ v16bf load_frag_a(const __bf16* base, int half) {
  v8bf lo = *reinterpret_cast<const v8bf*>(base + half * 16);
  v8bf hi = *reinterpret_cast<const v8bf*>(base + half * 16 + 8);
  v16bf r;
#pragma unroll
  for (int i = 0; i < 8; ++i) { r[i] = lo[i]; r[i + 8] = hi[i]; }
  return r;
}

__device__ __forceinline__ v16bf load_frag_b(const __bf16* base) {
  const v8bf* p = reinterpret_cast<const v8bf*>(base);
  v8bf lo = p[0], hi = p[1];
  v16bf r;
#pragma unroll
  for (int i = 0; i < 8; ++i) { r[i] = lo[i]; r[i + 8] = hi[i]; }
  return r;
}

// ---------------------------------------------------------------------------
// Kernel 1: fp32 -> bf16 elementwise convert (hidden_states)
// ---------------------------------------------------------------------------
__global__ __launch_bounds__(256) void k_f32_to_bf16(const float* __restrict__ in,
                                                     __bf16* __restrict__ out, int n) {
  int i = blockIdx.x * 256 + threadIdx.x;
  if (i < n) out[i] = (__bf16)in[i];
}

// ---------------------------------------------------------------------------
// Kernel 2: transpose W [K,N] -> Wt [N,K] with bf16 convert, 32x32 LDS tile
// ---------------------------------------------------------------------------
__global__ __launch_bounds__(256) void k_transpose_w(const float* __restrict__ Wq,
                                                     const float* __restrict__ Wk,
                                                     __bf16* __restrict__ WqT,
                                                     __bf16* __restrict__ WkT) {
  __shared__ float tile[32][33];
  const float* W = (blockIdx.z == 0) ? Wq : Wk;
  __bf16* Wt = (blockIdx.z == 0) ? WqT : WkT;
  const int N = D_DIM;
  int bx = blockIdx.x * 32, by = blockIdx.y * 32;
  int tx = threadIdx.x & 31, tg = threadIdx.x >> 5;  // 8 row-groups
#pragma unroll
  for (int r = tg; r < 32; r += 8) tile[r][tx] = W[(size_t)(by + r) * N + bx + tx];
  __syncthreads();
#pragma unroll
  for (int r = tg; r < 32; r += 8) Wt[(size_t)(bx + r) * N + by + tx] = (__bf16)tile[tx][r];
}

// ---------------------------------------------------------------------------
// Kernel 3: projection GEMM  Out = A[4096,512] @ W[512,512]  (bf16 in/out, f32
// accumulate). One 16x16 tile per wave, K-loop of 32 via wmma bf16.
// blockIdx.y selects (WqT -> Q) or (WkT -> K).
// ---------------------------------------------------------------------------
__global__ __launch_bounds__(256) void k_proj_gemm(const __bf16* __restrict__ A,
                                                   const __bf16* __restrict__ WqT,
                                                   const __bf16* __restrict__ WkT,
                                                   __bf16* __restrict__ Qo,
                                                   __bf16* __restrict__ Ko) {
  const int N = D_DIM, K = D_DIM;
  const __bf16* Wt = (blockIdx.y == 0) ? WqT : WkT;
  __bf16* Out = (blockIdx.y == 0) ? Qo : Ko;
  int wave = blockIdx.x * 8 + (threadIdx.x >> 5);
  int lane = threadIdx.x & 31;
  const int tiles_n = N / 16;
  int tm = wave / tiles_n, tn = wave % tiles_n;
  int r = lane & 15, half = lane >> 4;

  const __bf16* arow = A + (size_t)(tm * 16 + r) * K;   // A row (M = r)
  const __bf16* brow = Wt + (size_t)(tn * 16 + r) * K;  // W column (N = r)
  v8f acc = {};
  for (int k0 = 0; k0 < K; k0 += 32) {
    __builtin_prefetch(arow + k0 + 32, 0, 0);
    __builtin_prefetch(brow + k0 + 32, 0, 0);
    v16bf a = load_frag_a(arow + k0, half);
    v16bf b = load_frag_b(brow + k0 + half * 16);
    acc = __builtin_amdgcn_wmma_f32_16x16x32_bf16(false, a, false, b, (short)0, acc,
                                                  false, false);
  }
#pragma unroll
  for (int j = 0; j < 8; ++j) {
    int m = tm * 16 + half * 8 + j;
    Out[(size_t)m * N + tn * 16 + r] = (__bf16)acc[j];
  }
}

// ---------------------------------------------------------------------------
// Kernel 4: scores + online softmax stats + band score capture.
// Block = 8 waves sharing one (b,h), covering 128 consecutive query rows.
// The K j-tile stream (16x64 bf16 = 2 KB per tile) is staged cooperatively
// into a double-buffered LDS tile with global_load_async_to_lds_b64 (8 B per
// lane), overlapped with the WMMA compute of the current tile; per-iteration
// s_wait_asynccnt + barrier closes the pipeline. B-fragments then read from
// LDS (ds_load_b128). Per-row online (max, sumexp) in registers; raw band
// scores (|i-j|<=7) spilled for the context kernel.
// ---------------------------------------------------------------------------
__global__ __launch_bounds__(256) void k_scores(const __bf16* __restrict__ Q,
                                                const __bf16* __restrict__ Kb,
                                                float* __restrict__ bandS,
                                                float* __restrict__ rowMax,
                                                float* __restrict__ rowZ) {
  const int S = S_LEN, D = D_DIM, dh = DH_DIM;
  __shared__ __bf16 kbuf[2][16][DH_DIM];  // 2 x 2KB double buffer

  int tid = threadIdx.x;
  int wv = tid >> 5, lane = tid & 31;
  int blk = blockIdx.x;       // [0, B*H*16)
  int bh = blk >> 4;          // b*8 + h
  int big = blk & 15;         // which 128-row group
  int b = bh >> 3, h = bh & 7;
  int i0 = big * 128 + wv * 16;
  int r = lane & 15, half = lane >> 4;

  // Loop-invariant Q fragments (dh = 64 = two K-steps of 32)
  const __bf16* qrow = Q + ((size_t)b * S + i0 + r) * D + h * dh;
  v16bf a0 = load_frag_a(qrow, half);
  v16bf a1 = load_frag_a(qrow + 32, half);

  // Cooperative staging geometry: 16 rows x 128 bytes, 8 bytes per thread.
  const uint8_t* kgbase = (const uint8_t*)(Kb + ((size_t)b * S) * D + h * dh);
  int srow = tid >> 4;            // 0..15
  int scol = (tid & 15) * 8;      // byte offset within 128B row

  // Prologue: stage tile 0 into buffer 0
  {
    uint64_t g = (uint64_t)(uintptr_t)(kgbase + (size_t)srow * (D * 2) + scol);
    uint32_t l = (uint32_t)(uintptr_t)(&kbuf[0][srow][0]) + (uint32_t)scol;
    asm volatile("global_load_async_to_lds_b64 %0, %1, off"
                 :: "v"(l), "v"(g) : "memory");
    asm volatile("s_wait_asynccnt 0x0" ::: "memory");
  }
  __syncthreads();

  float rm[8], rs[8];
#pragma unroll
  for (int p = 0; p < 8; ++p) { rm[p] = -1e30f; rs[p] = 0.0f; }

  float* bandRow = bandS + (size_t)bh * S * 16;

  for (int tj = 0; tj < S / 16; ++tj) {
    int buf = tj & 1;
    // Kick off async staging of the next K tile into the other buffer.
    if (tj + 1 < S / 16) {
      uint64_t g = (uint64_t)(uintptr_t)(kgbase +
                     ((size_t)(tj + 1) * 16 + srow) * (D * 2) + scol);
      uint32_t l = (uint32_t)(uintptr_t)(&kbuf[buf ^ 1][srow][0]) + (uint32_t)scol;
      asm volatile("global_load_async_to_lds_b64 %0, %1, off"
                   :: "v"(l), "v"(g) : "memory");
    }

    // Compute current tile from LDS: B column c = r, K contiguous in the row.
    const __bf16* krow = &kbuf[buf][r][0];
    v8f acc = {};
    v16bf b0 = load_frag_b(krow + half * 16);
    acc = __builtin_amdgcn_wmma_f32_16x16x32_bf16(false, a0, false, b0, (short)0, acc,
                                                  false, false);
    v16bf b1 = load_frag_b(krow + 32 + half * 16);
    acc = __builtin_amdgcn_wmma_f32_16x16x32_bf16(false, a1, false, b1, (short)0, acc,
                                                  false, false);

    int jg = tj * 16 + r;  // global key column this lane owns
#pragma unroll
    for (int p = 0; p < 8; ++p) {
      float s = acc[p] * 0.125f;  // 1/sqrt(dh)
      int i = i0 + half * 8 + p;
      int d = jg - i;
      if (d >= -LOCAL_R && d <= LOCAL_R) bandRow[(size_t)i * 16 + (d + LOCAL_R)] = s;
      float nm = fmaxf(rm[p], s);
      rs[p] = rs[p] * __expf(rm[p] - nm) + __expf(s - nm);
      rm[p] = nm;
    }

    // Close the software pipeline: staged tile complete before next iteration.
    asm volatile("s_wait_asynccnt 0x0" ::: "memory");
    __syncthreads();
  }

  // Reduce (max,sum) across the 16 lanes holding each row's columns.
  for (int off = 1; off < 16; off <<= 1) {
#pragma unroll
    for (int p = 0; p < 8; ++p) {
      float om = __shfl_xor(rm[p], off);
      float os = __shfl_xor(rs[p], off);
      float nm = fmaxf(rm[p], om);
      rs[p] = rs[p] * __expf(rm[p] - nm) + os * __expf(om - nm);
      rm[p] = nm;
    }
  }
  if (r == 0) {
#pragma unroll
    for (int p = 0; p < 8; ++p) {
      int i = i0 + half * 8 + p;
      rowMax[(size_t)bh * S + i] = rm[p];
      rowZ[(size_t)bh * S + i] = rs[p];
    }
  }
}

// ---------------------------------------------------------------------------
// Kernel 5: band-sparse context. One block per (b,h,i); 15-tap weighted sum of
// hidden_states rows; local_w = exp(s-max) / (sum_band exp(s-max) + 1e-6 * Z).
// ---------------------------------------------------------------------------
__global__ __launch_bounds__(256) void k_context(const float* __restrict__ hs,
                                                 const float* __restrict__ bandS,
                                                 const float* __restrict__ rowMax,
                                                 const float* __restrict__ rowZ,
                                                 float* __restrict__ out) {
  const int S = S_LEN, D = D_DIM;
  int blk = blockIdx.x;  // [0, B*H*S)
  int i = blk & (S - 1);
  int bh = blk >> 11;    // b*8 + h
  int b = bh >> 3;
  __shared__ float w[16];
  __shared__ float invden;
  int tid = threadIdx.x;
  if (tid == 0) {
    float m = rowMax[(size_t)bh * S + i];
    float Z = rowZ[(size_t)bh * S + i];
    const float* srow = bandS + ((size_t)bh * S + i) * 16;
    float den = 1e-6f * Z;
#pragma unroll
    for (int j = 0; j < 2 * LOCAL_R + 1; ++j) {
      int jg = i - LOCAL_R + j;
      float v = 0.0f;
      if (jg >= 0 && jg < S) v = __expf(srow[j] - m);
      w[j] = v;
      den += v;
    }
    invden = 1.0f / den;
  }
  __syncthreads();
  const float* hsb = hs + (size_t)b * S * D;
  float acc0 = 0.0f, acc1 = 0.0f;
  int d0 = tid, d1 = tid + 256;
#pragma unroll
  for (int j = 0; j < 2 * LOCAL_R + 1; ++j) {
    int jg = i - LOCAL_R + j;
    if (jg >= 0 && jg < S) {
      float wj = w[j];
      acc0 += wj * hsb[(size_t)jg * D + d0];
      acc1 += wj * hsb[(size_t)jg * D + d1];
    }
  }
  float sc = invden;
  out[((size_t)bh * S + i) * D + d0] = acc0 * sc;
  out[((size_t)bh * S + i) * D + d1] = acc1 * sc;
}

// ---------------------------------------------------------------------------
extern "C" void kernel_launch(void* const* d_in, const int* in_sizes, int n_in,
                              void* d_out, int out_size, void* d_ws, size_t ws_size,
                              hipStream_t stream) {
  const float* hs = (const float*)d_in[0];  // [2,2048,512]
  const float* Wq = (const float*)d_in[1];  // [512,512]
  const float* Wk = (const float*)d_in[2];  // [512,512]
  float* out = (float*)d_out;               // [2,8,2048,512]

  char* ws = (char*)d_ws;
  const size_t NM = (size_t)B_NUM * S_LEN;  // 4096 rows
  __bf16* hsb = (__bf16*)(ws);                                // 4 MB
  __bf16* wqt = (__bf16*)(ws + (4u << 20));                   // 512 KB
  __bf16* wkt = (__bf16*)(ws + (4u << 20) + (512u << 10));    // 512 KB
  __bf16* Qb  = (__bf16*)(ws + (5u << 20));                   // 4 MB
  __bf16* Kb  = (__bf16*)(ws + (9u << 20));                   // 4 MB
  float* bandS  = (float*)(ws + (13u << 20));                 // 4 MB
  float* rowMax = (float*)(ws + (17u << 20));                 // 128 KB
  float* rowZ   = (float*)(ws + (17u << 20) + (128u << 10));  // 128 KB

  // 1) convert hidden_states to bf16
  {
    int n = (int)(NM * D_DIM);
    k_f32_to_bf16<<<(n + 255) / 256, 256, 0, stream>>>(hs, hsb, n);
  }
  // 2) transpose + convert Wq, Wk  ([K,N] -> [N,K] bf16)
  k_transpose_w<<<dim3(D_DIM / 32, D_DIM / 32, 2), 256, 0, stream>>>(Wq, Wk, wqt, wkt);
  // 3) Q = hsb @ Wq, K = hsb @ Wk   (bf16 WMMA, f32 accum)
  {
    int waves = (int)(NM / 16) * (D_DIM / 16);  // 8192
    k_proj_gemm<<<dim3(waves / 8, 2), 256, 0, stream>>>(hsb, wqt, wkt, Qb, Kb);
  }
  // 4) scores + softmax stats + band capture (async LDS staged bf16 WMMA)
  {
    int blocks = B_NUM * H_NUM * (S_LEN / 128);  // 256
    k_scores<<<blocks, 256, 0, stream>>>(Qb, Kb, bandS, rowMax, rowZ);
  }
  // 5) band-sparse context against fp32 hidden_states
  k_context<<<B_NUM * H_NUM * S_LEN, 256, 0, stream>>>(hs, bandS, rowMax, rowZ, out);
}
